// CEBlock_66632122630158
// MI455X (gfx1250) — compile-verified, matrix operand order checked by reading, not run on hardware
//
#include <hip/hip_runtime.h>
#include <hip/hip_bf16.h>
#include <math.h>

typedef __attribute__((ext_vector_type(16))) _Float16 v16h;
typedef __attribute__((ext_vector_type(8)))  _Float16 v8h;
typedef __attribute__((ext_vector_type(8)))  float    v8f;
typedef __attribute__((ext_vector_type(4)))  float    v4f;

#define BB   32
#define NN   320
#define LT_  64
#define LS_  256
#define CC   768
#define HH   12
#define DD   64
#define FF   3072
#define KEEP 180
#define REMV 76
#define NTOK 244           // LT + KEEP
#define BN   (BB*NN)       // 10240 rows
#define BT   (BB*NTOK)     // 7808 rows

// ---------------------------------------------------------------------------
// Register-blocked WMMA: each wave computes an (MT*16) x (NT*16) output block.
// A-fragment per ISA 16-bit layout: lane L -> row M = L%16; lanes 0-15 hold
// K = [0..7, 16..23], lanes 16-31 hold K = [8..15, 24..31] per 32-wide k-step
// (two contiguous 16B chunks -> b128 loads). B mirrored on the N dimension.
// Per k-step: MT*2 + NT*2 b128 loads feed MT*NT v_wmma ops.
// ---------------------------------------------------------------------------
template <int MT, int NT>
__device__ __forceinline__ void wmma_block_f16(const _Float16* __restrict__ A, int lda,
                                               const _Float16* __restrict__ W, int ldw,
                                               int K, int row0, int col0,
                                               v8f (&acc)[MT][NT]) {
  const int lane = threadIdx.x & 31;
  const int g    = lane >> 4;
  const _Float16* ar[MT];
  const _Float16* wr[NT];
#pragma unroll
  for (int i = 0; i < MT; ++i)
    ar[i] = A + (size_t)(row0 + i * 16 + (lane & 15)) * lda + 8 * g;
#pragma unroll
  for (int j = 0; j < NT; ++j)
    wr[j] = W + (size_t)(col0 + j * 16 + (lane & 15)) * ldw + 8 * g;
  for (int k = 0; k < K; k += 32) {
    v16h a[MT], b[NT];
#pragma unroll
    for (int i = 0; i < MT; ++i) {
      v8h a0 = *(const v8h*)(ar[i] + k);
      v8h a1 = *(const v8h*)(ar[i] + k + 16);
#pragma unroll
      for (int e = 0; e < 8; ++e) { a[i][e] = a0[e]; a[i][e + 8] = a1[e]; }
    }
#pragma unroll
    for (int j = 0; j < NT; ++j) {
      v8h b0 = *(const v8h*)(wr[j] + k);
      v8h b1 = *(const v8h*)(wr[j] + k + 16);
#pragma unroll
      for (int e = 0; e < 8; ++e) { b[j][e] = b0[e]; b[j][e + 8] = b1[e]; }
    }
#pragma unroll
    for (int i = 0; i < MT; ++i)
#pragma unroll
      for (int j = 0; j < NT; ++j)
        acc[i][j] = __builtin_amdgcn_wmma_f32_16x16x32_f16(
            false, a[i], false, b[j], (short)0, acc[i][j], false, false);
  }
}

// ---------------- LayerNorm (row-wise, C=768) -> f16 ------------------------
__global__ __launch_bounds__(256) void ln_f16_kernel(const float* __restrict__ x,
                                                     const float* __restrict__ gw,
                                                     const float* __restrict__ bw,
                                                     _Float16* __restrict__ out) {
  const int row = blockIdx.x;
  const int t   = threadIdx.x;
  const float* xr = x + (size_t)row * CC;
  __shared__ float red[256];
  float v[3]; float s = 0.f;
#pragma unroll
  for (int i = 0; i < 3; ++i) { v[i] = xr[t + 256 * i]; s += v[i]; }
  red[t] = s; __syncthreads();
  for (int o = 128; o > 0; o >>= 1) { if (t < o) red[t] += red[t + o]; __syncthreads(); }
  const float mu = red[0] * (1.0f / CC);
  __syncthreads();
  s = 0.f;
#pragma unroll
  for (int i = 0; i < 3; ++i) { float d = v[i] - mu; s += d * d; }
  red[t] = s; __syncthreads();
  for (int o = 128; o > 0; o >>= 1) { if (t < o) red[t] += red[t + o]; __syncthreads(); }
  const float rs = rsqrtf(red[0] * (1.0f / CC) + 1e-5f);
#pragma unroll
  for (int i = 0; i < 3; ++i) {
    int c = t + 256 * i;
    out[(size_t)row * CC + c] = (_Float16)((v[i] - mu) * rs * gw[c] + bw[c]);
  }
}

// ---------------- QKV GEMM: (BN x 2304) = xn @ qkv_w^T + b ------------------
// Scatter epilogue: q,k as (B,H,N,D) f16 (q pre-scaled by D^-0.5);
// v stored TRANSPOSED as (B,H,D,N) f16 so the AV GEMM B-operand is contiguous.
__global__ __launch_bounds__(256) void gemm_qkv_kernel(const _Float16* __restrict__ xn,
                                                       const _Float16* __restrict__ w,
                                                       const float* __restrict__ bias,
                                                       _Float16* __restrict__ q16,
                                                       _Float16* __restrict__ k16,
                                                       _Float16* __restrict__ vt16) {
  const int wid = (blockIdx.x * blockDim.x + threadIdx.x) >> 5;
  const int ngroups = (3 * CC) / 32;  // 72
  const int mg = wid / ngroups, ng = wid % ngroups;
  if (mg >= BN / 64) return;
  v8f acc[4][2] = {};
  wmma_block_f16<4, 2>(xn, CC, w, CC, CC, mg * 64, ng * 32, acc);
  const int lane = threadIdx.x & 31;
  const int g    = lane >> 4;
#pragma unroll
  for (int j = 0; j < 2; ++j) {
    const int c     = ng * 32 + j * 16 + (lane & 15);
    const int which = c / CC;
    const int rem   = c % CC;
    const int h = rem / DD, d = rem % DD;
    const float sc = (which == 0) ? 0.125f : 1.0f;  // D^-0.5 folded into q
    const float bi = bias[c];
#pragma unroll
    for (int i = 0; i < 4; ++i) {
#pragma unroll
      for (int r = 0; r < 8; ++r) {
        int m = mg * 64 + i * 16 + r + 8 * g;
        int b = m / NN, tok = m % NN;
        _Float16 val = (_Float16)((acc[i][j][r] + bi) * sc);
        if (which == 0)
          q16[(((size_t)b * HH + h) * NN + tok) * DD + d] = val;
        else if (which == 1)
          k16[(((size_t)b * HH + h) * NN + tok) * DD + d] = val;
        else
          vt16[(((size_t)b * HH + h) * DD + d) * NN + tok] = val;
      }
    }
  }
}

// ---------------- Attention logits: S = q @ k^T per (b,h) ------------------
__global__ __launch_bounds__(256) void gemm_qk_kernel(const _Float16* __restrict__ q16,
                                                      const _Float16* __restrict__ k16,
                                                      float* __restrict__ attn) {
  const int wid = (blockIdx.x * blockDim.x + threadIdx.x) >> 5;
  if (wid >= BB * HH * 50) return;   // per (b,h): 5 m-groups x 10 n-groups
  const int bh = wid / 50;
  const int t  = wid % 50;
  const int mg = t / 10, ng = t % 10;
  const _Float16* qb = q16 + (size_t)bh * NN * DD;
  const _Float16* kb = k16 + (size_t)bh * NN * DD;
  v8f acc[4][2] = {};
  wmma_block_f16<4, 2>(qb, DD, kb, DD, DD, mg * 64, ng * 32, acc);
  const int lane = threadIdx.x & 31;
  const int g    = lane >> 4;
  float* ab = attn + (size_t)bh * NN * NN;
#pragma unroll
  for (int i = 0; i < 4; ++i)
#pragma unroll
    for (int j = 0; j < 2; ++j) {
      const int n = ng * 32 + j * 16 + (lane & 15);
#pragma unroll
      for (int r = 0; r < 8; ++r) {
        int m = mg * 64 + i * 16 + r + 8 * g;
        ab[(size_t)m * NN + n] = acc[i][j][r];
      }
    }
}

// ---------------- Row softmax over 320, one wave per row, in place ---------
__global__ __launch_bounds__(256) void softmax_kernel(float* __restrict__ attn) {
  const int row = (blockIdx.x * blockDim.x + threadIdx.x) >> 5;
  if (row >= BB * HH * NN) return;
  const int lane = threadIdx.x & 31;
  float* p = attn + (size_t)row * NN;
  float v[10]; float mx = -3.0e38f;
#pragma unroll
  for (int i = 0; i < 10; ++i) { v[i] = p[lane + 32 * i]; mx = fmaxf(mx, v[i]); }
  for (int o = 16; o > 0; o >>= 1) mx = fmaxf(mx, __shfl_xor(mx, o, 32));
  float s = 0.f;
#pragma unroll
  for (int i = 0; i < 10; ++i) { v[i] = expf(v[i] - mx); s += v[i]; }
  for (int o = 16; o > 0; o >>= 1) s += __shfl_xor(s, o, 32);
  const float inv = 1.0f / s;
#pragma unroll
  for (int i = 0; i < 10; ++i) p[lane + 32 * i] = v[i] * inv;
}

// ---------------- attn_t = mean over heads & template rows -----------------
__global__ __launch_bounds__(256) void attn_t_kernel(const float* __restrict__ attn,
                                                     float* __restrict__ at) {
  const int b = blockIdx.x, j = threadIdx.x;  // 256 search columns
  float s = 0.f;
  for (int h = 0; h < HH; ++h) {
    const float* base = attn + (((size_t)b * HH + h) * NN) * NN + LT_ + j;
    for (int r = 0; r < LT_; ++r) s += base[(size_t)r * NN];
  }
  at[b * LS_ + j] = s * (1.0f / (HH * LT_));
}

// ---------------- Per-batch bitonic argsort (desc, stable ties) ------------
__global__ __launch_bounds__(256) void topk_kernel(const float* __restrict__ at,
                                                   const int* __restrict__ gis,
                                                   float* __restrict__ keep,
                                                   float* __restrict__ removed,
                                                   float* __restrict__ topkf,
                                                   int* __restrict__ topki) {
  __shared__ float sv[LS_];
  __shared__ int   si[LS_];
  const int b = blockIdx.x, t = threadIdx.x;
  sv[t] = at[b * LS_ + t];
  si[t] = t;
  __syncthreads();
  for (int ksz = 2; ksz <= LS_; ksz <<= 1) {
    for (int j = ksz >> 1; j > 0; j >>= 1) {
      int o = t ^ j;
      if (o > t) {
        float v0 = sv[t], v1 = sv[o];
        int   i0 = si[t], i1 = si[o];
        bool desc = ((t & ksz) == 0);
        bool inorder = (v0 > v1) || (v0 == v1 && i0 < i1);
        if (desc != inorder) { sv[t] = v1; sv[o] = v0; si[t] = i1; si[o] = i0; }
      }
      __syncthreads();
    }
  }
  const int sidx = si[t];
  if (t < KEEP) {
    keep[b * KEEP + t]  = (float)gis[b * LS_ + sidx];
    topkf[b * KEEP + t] = (float)sidx;
    topki[b * KEEP + t] = sidx;
  } else {
    removed[b * REMV + (t - KEEP)] = (float)gis[b * LS_ + sidx];
  }
}

// ---------------- out = attn(f32->f16) @ v per (b,h) -----------------------
// v stored transposed (B,H,D,N) so B-fragments are contiguous b128 loads.
__global__ __launch_bounds__(256) void gemm_av_kernel(const float* __restrict__ attn,
                                                      const _Float16* __restrict__ vt16,
                                                      _Float16* __restrict__ out16) {
  const int wid = (blockIdx.x * blockDim.x + threadIdx.x) >> 5;
  if (wid >= BB * HH * 10) return;   // per (b,h): 5 m-groups x 2 n-groups
  const int bh = wid / 10;
  const int t  = wid % 10;
  const int mg = t / 2, ng = t % 2;
  const int lane = threadIdx.x & 31;
  const int g    = lane >> 4;
  const float* ar[4];
  const _Float16* wr[2];
#pragma unroll
  for (int i = 0; i < 4; ++i)
    ar[i] = attn + ((size_t)bh * NN + mg * 64 + i * 16 + (lane & 15)) * NN + 8 * g;
#pragma unroll
  for (int j = 0; j < 2; ++j)
    wr[j] = vt16 + ((size_t)bh * DD + ng * 32 + j * 16 + (lane & 15)) * NN + 8 * g;
  v8f acc[4][2] = {};
  for (int k = 0; k < NN; k += 32) {
    v16h a[4], b[2];
#pragma unroll
    for (int i = 0; i < 4; ++i) {
      const float* p = ar[i] + k;
      v4f f0 = ((const v4f*)p)[0];
      v4f f1 = ((const v4f*)p)[1];
      v4f f2 = ((const v4f*)(p + 16))[0];
      v4f f3 = ((const v4f*)(p + 16))[1];
#pragma unroll
      for (int e = 0; e < 4; ++e) {
        a[i][e]      = (_Float16)f0[e];
        a[i][4 + e]  = (_Float16)f1[e];
        a[i][8 + e]  = (_Float16)f2[e];
        a[i][12 + e] = (_Float16)f3[e];
      }
    }
#pragma unroll
    for (int j = 0; j < 2; ++j) {
      v8h b0 = *(const v8h*)(wr[j] + k);
      v8h b1 = *(const v8h*)(wr[j] + k + 16);
#pragma unroll
      for (int e = 0; e < 8; ++e) { b[j][e] = b0[e]; b[j][e + 8] = b1[e]; }
    }
#pragma unroll
    for (int i = 0; i < 4; ++i)
#pragma unroll
      for (int j = 0; j < 2; ++j)
        acc[i][j] = __builtin_amdgcn_wmma_f32_16x16x32_f16(
            false, a[i], false, b[j], (short)0, acc[i][j], false, false);
  }
  const int b_ = bh / HH, h = bh % HH;
#pragma unroll
  for (int i = 0; i < 4; ++i)
#pragma unroll
    for (int j = 0; j < 2; ++j) {
      const int n = ng * 32 + j * 16 + (lane & 15);
#pragma unroll
      for (int r = 0; r < 8; ++r) {
        int mm = mg * 64 + i * 16 + r + 8 * g;
        out16[((size_t)b_ * NN + mm) * CC + h * DD + n] = (_Float16)acc[i][j][r];
      }
    }
}

// ---------------- Generic GEMM + bias + residual (f32 out) -----------------
__global__ __launch_bounds__(256) void gemm_bias_res_kernel(const _Float16* __restrict__ A, int lda,
                                                            const _Float16* __restrict__ W, int ldw, int K,
                                                            const float* __restrict__ bias,
                                                            const float* __restrict__ resid,
                                                            float* __restrict__ out,
                                                            int ncols, int mgroups) {
  const int wid = (blockIdx.x * blockDim.x + threadIdx.x) >> 5;
  const int ngroups = ncols >> 5;
  const int mg = wid / ngroups, ng = wid % ngroups;
  if (mg >= mgroups) return;
  v8f acc[4][2] = {};
  wmma_block_f16<4, 2>(A, lda, W, ldw, K, mg * 64, ng * 32, acc);
  const int lane = threadIdx.x & 31;
  const int g    = lane >> 4;
#pragma unroll
  for (int j = 0; j < 2; ++j) {
    const int c = ng * 32 + j * 16 + (lane & 15);
    const float bi = bias[c];
#pragma unroll
    for (int i = 0; i < 4; ++i)
#pragma unroll
      for (int r = 0; r < 8; ++r) {
        size_t idx = (size_t)(mg * 64 + i * 16 + r + 8 * g) * ncols + c;
        out[idx] = resid[idx] + acc[i][j][r] + bi;
      }
  }
}

// ---------------- FC1 GEMM + bias + exact GELU -> f16 ----------------------
__global__ __launch_bounds__(256) void gemm_gelu_kernel(const _Float16* __restrict__ A,
                                                        const _Float16* __restrict__ W,
                                                        const float* __restrict__ bias,
                                                        _Float16* __restrict__ out, int mgroups) {
  const int wid = (blockIdx.x * blockDim.x + threadIdx.x) >> 5;
  const int ngroups = FF >> 5;  // 96
  const int mg = wid / ngroups, ng = wid % ngroups;
  if (mg >= mgroups) return;
  v8f acc[4][2] = {};
  wmma_block_f16<4, 2>(A, CC, W, CC, CC, mg * 64, ng * 32, acc);
  const int lane = threadIdx.x & 31;
  const int g    = lane >> 4;
#pragma unroll
  for (int j = 0; j < 2; ++j) {
    const int c = ng * 32 + j * 16 + (lane & 15);
    const float bi = bias[c];
#pragma unroll
    for (int i = 0; i < 4; ++i)
#pragma unroll
      for (int r = 0; r < 8; ++r) {
        float xx = acc[i][j][r] + bi;
        float ge = 0.5f * xx * (1.0f + erff(xx * 0.70710678118654752f));
        out[(size_t)(mg * 64 + i * 16 + r + 8 * g) * FF + c] = (_Float16)ge;
      }
  }
}

// ---------------- Token gather: tokens_new = [x[:64], x[64+topk]] ----------
__global__ __launch_bounds__(256) void gather_tokens_kernel(const float* __restrict__ xpost,
                                                            const int* __restrict__ topki,
                                                            float* __restrict__ tokens) {
  size_t i = (size_t)blockIdx.x * blockDim.x + threadIdx.x;
  if (i >= (size_t)BT * CC) return;
  int c   = (int)(i % CC);
  int row = (int)(i / CC);
  int b = row / NTOK, r = row % NTOK;
  int src = (r < LT_) ? r : (LT_ + topki[b * KEEP + (r - LT_)]);
  tokens[i] = xpost[((size_t)b * NN + src) * CC + c];
}

// ---------------- Utility conversions/copies -------------------------------
__global__ __launch_bounds__(256) void f32_to_f16_kernel(const float* __restrict__ a,
                                                         _Float16* __restrict__ o, size_t n) {
  size_t i = (size_t)blockIdx.x * blockDim.x + threadIdx.x;
  if (i < n) o[i] = (_Float16)a[i];
}
__global__ __launch_bounds__(256) void copy_f32_kernel(const float* __restrict__ a,
                                                       float* __restrict__ o, size_t n) {
  size_t i = (size_t)blockIdx.x * blockDim.x + threadIdx.x;
  if (i < n) o[i] = a[i];
}
__global__ __launch_bounds__(256) void i32_to_f32_kernel(const int* __restrict__ a,
                                                         float* __restrict__ o, int n) {
  int i = blockIdx.x * blockDim.x + threadIdx.x;
  if (i < n) o[i] = (float)a[i];
}

// ===========================================================================
extern "C" void kernel_launch(void* const* d_in, const int* in_sizes, int n_in,
                              void* d_out, int out_size, void* d_ws, size_t ws_size,
                              hipStream_t stream) {
  (void)in_sizes; (void)n_in; (void)out_size; (void)ws_size;
  const float* x       = (const float*)d_in[0];
  const float* x_other = (const float*)d_in[1];
  const int*   git     = (const int*)d_in[2];
  const int*   gis     = (const int*)d_in[3];
  const float* n1g = (const float*)d_in[4];
  const float* n1b = (const float*)d_in[5];
  const float* qkv_w = (const float*)d_in[6];
  const float* qkv_b = (const float*)d_in[7];
  const float* proj_w = (const float*)d_in[8];
  const float* proj_b = (const float*)d_in[9];
  const float* n2g = (const float*)d_in[10];
  const float* n2b = (const float*)d_in[11];
  const float* n3g = (const float*)d_in[12];
  const float* n3b = (const float*)d_in[13];
  const float* fc1_w = (const float*)d_in[14];
  const float* fc1_b = (const float*)d_in[15];
  const float* fc2_w = (const float*)d_in[16];
  const float* fc2_b = (const float*)d_in[17];
  const float* fc1b_w = (const float*)d_in[18];
  const float* fc1b_b = (const float*)d_in[19];
  const float* fc2b_w = (const float*)d_in[20];
  const float* fc2b_b = (const float*)d_in[21];

  float* out = (float*)d_out;
  // Output layout (flat floats, reference return order)
  float* o_xnew = out;                     // 32*244*768
  float* o_xori = out + 5996544;           // 32*320*768
  float* o_xoth = out + 13860864;          // 32*320*768
  float* o_git  = out + 21725184;          // 32*64
  float* o_keep = out + 21727232;          // 32*180
  float* o_rem  = out + 21732992;          // 32*76
  float* o_attn = out + 21735424;          // 32*12*320*320 (softmaxed, in place)
  float* o_topk = out + 61057024;          // 32*180

  // Workspace carve-out (256B-aligned regions)
  char* ws = (char*)d_ws;
  size_t off = 0;
  auto alloc = [&](size_t bytes) -> char* {
    off = (off + 255) & ~(size_t)255;
    char* p = ws + off;
    off += bytes;
    return p;
  };
  _Float16* qkvw16  = (_Float16*)alloc((size_t)3 * CC * CC * 2);
  _Float16* projw16 = (_Float16*)alloc((size_t)CC * CC * 2);
  _Float16* fc1w16  = (_Float16*)alloc((size_t)FF * CC * 2);
  _Float16* fc2w16  = (_Float16*)alloc((size_t)CC * FF * 2);
  _Float16* fc1bw16 = (_Float16*)alloc((size_t)FF * CC * 2);
  _Float16* fc2bw16 = (_Float16*)alloc((size_t)CC * FF * 2);
  _Float16* a16  = (_Float16*)alloc((size_t)BN * CC * 2);           // LN out (reused)
  _Float16* q16  = (_Float16*)alloc((size_t)BB * HH * NN * DD * 2);
  _Float16* k16  = (_Float16*)alloc((size_t)BB * HH * NN * DD * 2);
  _Float16* vt16 = (_Float16*)alloc((size_t)BB * HH * DD * NN * 2); // transposed v
  _Float16* o16  = (_Float16*)alloc((size_t)BN * CC * 2);           // attn-out (B,N,C)
  float*  xpost  = (float*)alloc((size_t)BN * CC * 4);              // x + proj
  _Float16* h16  = (_Float16*)alloc((size_t)BN * FF * 2);           // MLP hidden (reused)
  float* tokens  = (float*)alloc((size_t)BT * CC * 4);
  _Float16* t16  = (_Float16*)alloc((size_t)BT * CC * 2);
  float*     at  = (float*)alloc((size_t)BB * LS_ * 4);
  int*    topki  = (int*)alloc((size_t)BB * KEEP * 4);

  auto cvt = [&](const float* src, _Float16* dst, size_t n) {
    f32_to_f16_kernel<<<(unsigned)((n + 255) / 256), 256, 0, stream>>>(src, dst, n);
  };
  // Weight conversions
  cvt(qkv_w,  qkvw16,  (size_t)3 * CC * CC);
  cvt(proj_w, projw16, (size_t)CC * CC);
  cvt(fc1_w,  fc1w16,  (size_t)FF * CC);
  cvt(fc2_w,  fc2w16,  (size_t)CC * FF);
  cvt(fc1b_w, fc1bw16, (size_t)FF * CC);
  cvt(fc2b_w, fc2bw16, (size_t)CC * FF);

  // 1) LN1 -> f16
  ln_f16_kernel<<<BN, 256, 0, stream>>>(x, n1g, n1b, a16);
  // 2) QKV GEMM (10240x2304x768): 160 m-groups x 72 n-groups of 64x32
  gemm_qkv_kernel<<<((BN / 64) * 72 + 7) / 8, 256, 0, stream>>>(a16, qkvw16, qkv_b, q16, k16, vt16);
  // 3) Attention logits per (b,h): 5x10 groups of 64x32
  gemm_qk_kernel<<<(BB * HH * 50 + 7) / 8, 256, 0, stream>>>(q16, k16, o_attn);
  // 4) Softmax rows in place (attn output is post-softmax, as reference)
  softmax_kernel<<<(BB * HH * NN + 7) / 8, 256, 0, stream>>>(o_attn);
  // 5) Token scores + top-k selection
  attn_t_kernel<<<BB, 256, 0, stream>>>(o_attn, at);
  topk_kernel<<<BB, 256, 0, stream>>>(at, gis, o_keep, o_rem, o_topk, topki);
  // 6) out = attn @ v (v transposed), scatter to (B,N,C) f16
  gemm_av_kernel<<<(BB * HH * 10 + 7) / 8, 256, 0, stream>>>(o_attn, vt16, o16);
  // 7) proj GEMM + residual: xpost = x + out@proj_w^T + b
  gemm_bias_res_kernel<<<((BN / 64) * (CC / 32) + 7) / 8, 256, 0, stream>>>(
      o16, CC, projw16, CC, CC, proj_b, x, xpost, CC, BN / 64);
  // 8) x_ori branch: LN3 -> fc1b(GELU) -> fc2b + residual
  ln_f16_kernel<<<BN, 256, 0, stream>>>(xpost, n3g, n3b, a16);
  gemm_gelu_kernel<<<((BN / 64) * (FF / 32) + 7) / 8, 256, 0, stream>>>(a16, fc1bw16, fc1b_b, h16, BN / 64);
  gemm_bias_res_kernel<<<((BN / 64) * (CC / 32) + 7) / 8, 256, 0, stream>>>(
      h16, FF, fc2bw16, FF, FF, fc2b_b, xpost, o_xori, CC, BN / 64);
  // 9) Pruned-token branch
  gather_tokens_kernel<<<(unsigned)(((size_t)BT * CC + 255) / 256), 256, 0, stream>>>(xpost, topki, tokens);
  ln_f16_kernel<<<BT, 256, 0, stream>>>(tokens, n2g, n2b, t16);
  gemm_gelu_kernel<<<((BT / 64) * (FF / 32) + 7) / 8, 256, 0, stream>>>(t16, fc1w16, fc1_b, h16, BT / 64);
  gemm_bias_res_kernel<<<((BT / 64) * (CC / 32) + 7) / 8, 256, 0, stream>>>(
      h16, FF, fc2w16, FF, FF, fc2_b, tokens, o_xnew, CC, BT / 64);
  // 10) Pass-through outputs
  copy_f32_kernel<<<(unsigned)(((size_t)BN * CC + 255) / 256), 256, 0, stream>>>(x_other, o_xoth, (size_t)BN * CC);
  i32_to_f32_kernel<<<(BB * LT_ + 255) / 256, 256, 0, stream>>>(git, o_git, BB * LT_);
}
